// OneHeadGraphAttetion_55688545960296
// MI455X (gfx1250) — matrix operand order repaired and checked by reading, not compiled, not run on hardware
//
#include <hip/hip_runtime.h>

#define NNODES 50000
#define NEDGES 800000
#define CIN    128
#define ATTRC  70
#define EDGEF  65

// ---------------- WMMA fragment types ----------------
typedef __attribute__((ext_vector_type(16))) __bf16 bf16x16;
typedef __attribute__((ext_vector_type(8)))  float  f32x8;
typedef unsigned int u32x4 __attribute__((ext_vector_type(4)));
typedef int          i32x8 __attribute__((ext_vector_type(8)));
typedef int          i32x4 __attribute__((ext_vector_type(4)));
union FragAB { bf16x16 v; unsigned int u[8]; };

__device__ __forceinline__ unsigned short f2bf(float f) {
  unsigned int u = __float_as_uint(f);
  u = u + 0x7FFFu + ((u >> 16) & 1u);            // round-to-nearest-even
  return (unsigned short)(u >> 16);
}
__device__ __forceinline__ unsigned int packbf2(float a, float b) {
  unsigned int ua = __float_as_uint(a);
  unsigned int ub = __float_as_uint(b);
  ua = (ua + 0x7FFFu + ((ua >> 16) & 1u)) >> 16;
  ub = (ub + 0x7FFFu + ((ub >> 16) & 1u)) & 0xFFFF0000u;
  return ua | ub;                                 // low half = first elem (little endian)
}

// A-layout (16-bit A 16x32, ISA 7.12.2): lane half hi, slot e -> K = hi*8 + e (+8 if e>=8).
// Dword v holds a consecutive K pair; uint index within a bf16-pair row:
//   v<4 : c*16 + hi*4 + v         v>=4 : c*16 + hi*4 + 8 + (v-4)
__device__ __forceinline__ bf16x16 load_a_frag(const unsigned int* __restrict__ row,
                                               int c, int hi) {
  FragAB a;
  const int bp = c * 16 + hi * 4;
#pragma unroll
  for (int v = 0; v < 4; ++v) { a.u[v] = row[bp + v]; a.u[4 + v] = row[bp + 8 + v]; }
  return a.v;
}
// B frags pre-swizzled: 8 contiguous dwords per (ntile, lane)
__device__ __forceinline__ bf16x16 load_b_frag(const unsigned int* __restrict__ sB,
                                               int nt, int lane) {
  FragAB b;
  const unsigned int* q = sB + (nt * 32 + lane) * 8;
#pragma unroll
  for (int v = 0; v < 8; ++v) b.u[v] = q[v];
  return b.v;
}

// 8-WMMA burst over one 32-K chunk with rotated B-fragment preload.
__device__ __forceinline__ void gemm_chunk(const unsigned int* __restrict__ sBc,
                                           const unsigned int* __restrict__ arow,
                                           int c, int hi, int lane,
                                           f32x8* __restrict__ acc) {
  bf16x16 a    = load_a_frag(arow, c, hi);
  bf16x16 bcur = load_b_frag(sBc, 0, lane);
#pragma unroll
  for (int nt = 0; nt < 8; ++nt) {
    bf16x16 bnxt = bcur;
    if (nt < 7) bnxt = load_b_frag(sBc, nt + 1, lane);
    acc[nt] = __builtin_amdgcn_wmma_f32_16x16x32_bf16(
        false, a, false, bcur, (short)0, acc[nt], false, false);
    bcur = bnxt;
  }
}

// ---- Tensor Data Mover: DMA one 8KB weight chunk (2048 dwords) global -> LDS ----
// D# per CDNA5 ISA ch.8: group0 {count=1, lds_addr, global_addr[56:0], type=2},
// group1 {data_size=4B, tensor_dim0=2048, tensor_dim1=1, tile_dim0=2048, tile_dim1=1,
//         tensor_dim0_stride=2048}; groups 2/3 zero (<=2D tile).
__device__ __forceinline__ void tdm_load_chunk(const void* gsrc, unsigned lds_off) {
  unsigned long long ga = (unsigned long long)gsrc;
  u32x4 g0;
  g0[0] = 1u;                                                  // count=1, user mode
  g0[1] = lds_off;                                             // lds_addr (bytes)
  g0[2] = (unsigned)(ga & 0xFFFFFFFFu);                        // global_addr[31:0]
  g0[3] = (unsigned)((ga >> 32) & 0x01FFFFFFu) | 0x80000000u;  // addr[56:32] | type=2
  i32x8 g1;
  g1[0] = 0x20000;                 // workgroup_mask=0, data_size=2 (4 bytes)
  g1[1] = (int)(2048u << 16);      // tensor_dim0[15:0] in bits 63:48
  g1[2] = (int)(1u << 16);         // tensor_dim0[31:16]=0 | tensor_dim1[15:0]=1
  g1[3] = (int)(2048u << 16);      // tensor_dim1[31:16]=0 | tile_dim0=2048
  g1[4] = 1;                       // tile_dim1=1, tile_dim2=0 (unused)
  g1[5] = 2048;                    // tensor_dim0_stride[31:0]
  g1[6] = 0;
  g1[7] = 0;
  i32x4 z4 = {0, 0, 0, 0};
  i32x8 z8 = {0, 0, 0, 0, 0, 0, 0, 0};
  __builtin_amdgcn_tensor_load_to_lds(g0, g1, z4, z4, z8, 0);
}
__device__ __forceinline__ unsigned lds_offset(const void* p) {
  return (unsigned)(unsigned long long)p;   // addrspace(3) pointer value == LDS offset
}

// ------------- weight packing (B-layout: lane l holds col n=l&15, K=(l>>4)*8+e (+8 if e>=8)) -------------
__device__ __forceinline__ void pack_one(const float* __restrict__ src,
                                         unsigned short* __restrict__ dst,
                                         int i, int Krows) {
  const int c    = i >> 12;        // / 4096 elems per 32-K chunk
  const int rem  = i & 4095;
  const int nt   = rem >> 9;       // / 512
  const int rem2 = rem & 511;
  const int lane = rem2 >> 4;
  const int e    = rem2 & 15;
  const int n    = nt * 16 + (lane & 15);
  const int k    = c * 32 + (lane >> 4) * 8 + e + ((e >> 3) << 3);
  const float v  = (k < Krows) ? src[k * CIN + n] : 0.0f;
  dst[i] = f2bf(v);
}

__global__ __launch_bounds__(256) void pack_weights_kernel(
    const float* __restrict__ Wf, const float* __restrict__ Wq,
    const float* __restrict__ Wk, const float* __restrict__ Wu,
    unsigned short* __restrict__ Wf_p, unsigned short* __restrict__ Wq_p,
    unsigned short* __restrict__ Wk_p, unsigned short* __restrict__ Wu_p) {
  int i = blockIdx.x * 256 + threadIdx.x;
  if (i < 15 * 4096) { pack_one(Wf, Wf_p, i, 449); return; }
  i -= 15 * 4096;
  if (i < 4 * 4096) { pack_one(Wq, Wq_p, i, 128); return; }
  i -= 4 * 4096;
  if (i < 4 * 4096) { pack_one(Wk, Wk_p, i, 128); return; }
  i -= 4 * 4096;
  if (i < 4 * 4096) { pack_one(Wu, Wu_p, i, 128); return; }
}

__global__ __launch_bounds__(256) void zero_f32(float* __restrict__ p, int n) {
  int i = blockIdx.x * 256 + threadIdx.x;
  if (i < n) p[i] = 0.0f;
}

// =================== Pass A: per-edge GEMMs + attention scalar ===================
// 4 waves / block, 16 edges / wave -> 64 edges / block, 12500 blocks.
__global__ __launch_bounds__(128) void edge_pass_a(
    const float* __restrict__ x, const int* __restrict__ ei,
    const float* __restrict__ ea,
    const unsigned short* __restrict__ Wf_p, const unsigned short* __restrict__ Wq_p,
    const unsigned short* __restrict__ Wk_p,
    const float* __restrict__ bfv, const float* __restrict__ Wa,
    unsigned int* __restrict__ f_ws, float* __restrict__ a_ws,
    float* __restrict__ a_sum) {
  __shared__ unsigned int sF[4][16][240];   // fcat rows, bf16 pairs (480 K padded)
  __shared__ unsigned int sB[2][2048];      // double-buffered 32-K weight chunk
  __shared__ unsigned int sO[4][16][64];    // f rows, bf16 pairs (128 cols)
  __shared__ float        sM[4][16];        // edge mask

  const int tid  = threadIdx.x;
  const int w    = tid >> 5;
  const int lane = tid & 31;
  const int hi   = lane >> 4;
  const int lo   = lane & 15;
  const int e0   = blockIdx.x * 64 + w * 16;

  // ---- build fcat = [ni | nj | ni-nj | edge_attr[:,0:65] | 0-pad] in bf16 ----
  for (int m = 0; m < 16; ++m) {
    const int eg = e0 + m;
    const int s  = ei[eg];
    const int t  = ei[NEDGES + eg];
    const float2* xs = (const float2*)(x + (size_t)s * CIN);
    const float2* xt = (const float2*)(x + (size_t)t * CIN);
    for (int j = lane; j < 64; j += 32) {
      float2 av = xs[j], bv = xt[j];
      sF[w][m][j]        = packbf2(av.x, av.y);
      sF[w][m][64 + j]   = packbf2(bv.x, bv.y);
      sF[w][m][128 + j]  = packbf2(av.x - bv.x, av.y - bv.y);
    }
    const float* arow = ea + (size_t)eg * ATTRC;
    for (int j = lane; j < 48; j += 32) {
      float v0 = (2 * j     < EDGEF) ? arow[2 * j]     : 0.0f;
      float v1 = (2 * j + 1 < EDGEF) ? arow[2 * j + 1] : 0.0f;
      sF[w][m][192 + j] = packbf2(v0, v1);
    }
    if (lane == 0) sM[w][m] = (arow[0] < 8.0f) ? 1.0f : 0.0f;
  }

  const f32x8 vzero = {0.f, 0.f, 0.f, 0.f, 0.f, 0.f, 0.f, 0.f};

  // ---- f = elu(fcat @ Wf + bf) * mask, K = 480 (15 chunks, TDM double-buffered) ----
  f32x8 facc[8];
#pragma unroll
  for (int i = 0; i < 8; ++i) facc[i] = vzero;
  __syncthreads();
  if (w == 0) tdm_load_chunk(Wf_p, lds_offset(&sB[0][0]));
  for (int c = 0; c < 15; ++c) {
    if (w == 0) __builtin_amdgcn_s_wait_tensorcnt(0);   // chunk c landed
    __syncthreads();                                    // publish; all readers past c-1
    if (w == 0 && c + 1 < 15)
      tdm_load_chunk(Wf_p + (size_t)(c + 1) * 4096, lds_offset(&sB[(c + 1) & 1][0]));
    gemm_chunk(sB[c & 1], &sF[w][lo][0], c, hi, lane, facc);
  }

  // bias + ELU + mask, pack f (bf16) into sO (C-layout: m = r + 8*hi, col = nt*16 + lo)
#pragma unroll
  for (int nt = 0; nt < 8; ++nt) {
    const int col = nt * 16 + lo;
    const float bc = bfv[col];
#pragma unroll
    for (int r = 0; r < 8; ++r) {
      const int rm = r + hi * 8;
      float v = facc[nt][r] + bc;
      v = (v > 0.0f) ? v : expm1f(v);
      v *= sM[w][rm];
      ((unsigned short*)&sO[w][rm][0])[col] = f2bf(v);
    }
  }
  // spill f rows to global (bf16, row-major, fully coalesced)
  for (int i = lane; i < 1024; i += 32)
    f_ws[(size_t)e0 * 64 + i] = sO[w][i >> 6][i & 63];

  // ---- q = ni @ Wq (A from fcat cols 0..127) ----
  f32x8 qacc[8], kacc[8];
#pragma unroll
  for (int i = 0; i < 8; ++i) { qacc[i] = vzero; kacc[i] = vzero; }
  __syncthreads();   // everyone done reading sB from the f-loop
  if (w == 0) tdm_load_chunk(Wq_p, lds_offset(&sB[0][0]));
  for (int c = 0; c < 4; ++c) {
    if (w == 0) __builtin_amdgcn_s_wait_tensorcnt(0);
    __syncthreads();
    if (w == 0 && c + 1 < 4)
      tdm_load_chunk(Wq_p + (size_t)(c + 1) * 4096, lds_offset(&sB[(c + 1) & 1][0]));
    gemm_chunk(sB[c & 1], &sF[w][lo][0], c, hi, lane, qacc);
  }
  // ---- k = f @ Wk (A from sO) ----
  __syncthreads();
  if (w == 0) tdm_load_chunk(Wk_p, lds_offset(&sB[0][0]));
  for (int c = 0; c < 4; ++c) {
    if (w == 0) __builtin_amdgcn_s_wait_tensorcnt(0);
    __syncthreads();
    if (w == 0 && c + 1 < 4)
      tdm_load_chunk(Wk_p + (size_t)(c + 1) * 4096, lds_offset(&sB[(c + 1) & 1][0]));
    gemm_chunk(sB[c & 1], &sO[w][lo][0], c, hi, lane, kacc);
  }

  // ---- a = tanh( sum_c q*k*scale*Wa[c] ), 16-lane butterfly reduction ----
  const float scale = 0.08838834764831845f;  // 1/sqrt(128)
  float p[8];
#pragma unroll
  for (int r = 0; r < 8; ++r) p[r] = 0.0f;
#pragma unroll
  for (int nt = 0; nt < 8; ++nt) {
    const float was = Wa[nt * 16 + lo] * scale;
#pragma unroll
    for (int r = 0; r < 8; ++r) p[r] += qacc[nt][r] * kacc[nt][r] * was;
  }
#pragma unroll
  for (int r = 0; r < 8; ++r) {
    float v = p[r];
    v += __shfl_xor(v, 1);
    v += __shfl_xor(v, 2);
    v += __shfl_xor(v, 4);
    v += __shfl_xor(v, 8);
    p[r] = v;
  }
  if (lo == 0) {
#pragma unroll
    for (int r = 0; r < 8; ++r) {
      const int eg = e0 + hi * 8 + r;
      const float av = tanhf(p[r]);
      a_ws[eg] = av;
      const int s = ei[eg];
      __hip_atomic_fetch_add(&a_sum[s], av, __ATOMIC_RELAXED, __HIP_MEMORY_SCOPE_AGENT);
    }
  }
}

// =================== Pass B: s = exp(a - a_sum[src]); scatter s*f ===================
// 1 wave per edge (4 channels per lane), 8 edges / block, 100000 blocks.
__global__ __launch_bounds__(256) void edge_pass_b(
    const int* __restrict__ ei, const unsigned int* __restrict__ f_ws,
    const float* __restrict__ a_ws, const float* __restrict__ a_sum,
    float* __restrict__ aggr) {
  const int eg   = blockIdx.x * 8 + (threadIdx.x >> 5);
  const int lane = threadIdx.x & 31;
  const int s    = ei[eg];
  const float sc = expf(a_ws[eg] - a_sum[s]);
  const unsigned int* frow = f_ws + (size_t)eg * 64;
  const unsigned int u0 = frow[lane * 2];
  const unsigned int u1 = frow[lane * 2 + 1];
  float* dst = aggr + (size_t)s * CIN + lane * 4;
  const float f0 = __uint_as_float(u0 << 16);
  const float f1 = __uint_as_float(u0 & 0xFFFF0000u);
  const float f2 = __uint_as_float(u1 << 16);
  const float f3 = __uint_as_float(u1 & 0xFFFF0000u);
  __hip_atomic_fetch_add(dst + 0, sc * f0, __ATOMIC_RELAXED, __HIP_MEMORY_SCOPE_AGENT);
  __hip_atomic_fetch_add(dst + 1, sc * f1, __ATOMIC_RELAXED, __HIP_MEMORY_SCOPE_AGENT);
  __hip_atomic_fetch_add(dst + 2, sc * f2, __ATOMIC_RELAXED, __HIP_MEMORY_SCOPE_AGENT);
  __hip_atomic_fetch_add(dst + 3, sc * f3, __ATOMIC_RELAXED, __HIP_MEMORY_SCOPE_AGENT);
}

// =================== Pass C: out = (x + aggr) @ Wu + bu ===================
__global__ __launch_bounds__(128) void node_pass(
    const float* __restrict__ x, const float* __restrict__ aggr,
    const unsigned short* __restrict__ Wu_p, const float* __restrict__ bu,
    float* __restrict__ out) {
  __shared__ unsigned int sA[4][16][64];
  __shared__ unsigned int sB[2][2048];
  const int tid  = threadIdx.x;
  const int w    = tid >> 5;
  const int lane = tid & 31;
  const int hi   = lane >> 4;
  const int lo   = lane & 15;
  const int n0   = blockIdx.x * 64 + w * 16;
  const bool active = (n0 < NNODES);  // tiles never partial (50000 % 16 == 0)

  if (active) {
    for (int m = 0; m < 16; ++m) {
      const size_t base = (size_t)(n0 + m) * CIN;
      for (int j = lane; j < 64; j += 32) {
        float v0 = x[base + 2 * j]     + aggr[base + 2 * j];
        float v1 = x[base + 2 * j + 1] + aggr[base + 2 * j + 1];
        sA[w][m][j] = packbf2(v0, v1);
      }
    }
  }
  const f32x8 vzero = {0.f, 0.f, 0.f, 0.f, 0.f, 0.f, 0.f, 0.f};
  f32x8 acc[8];
#pragma unroll
  for (int i = 0; i < 8; ++i) acc[i] = vzero;
  __syncthreads();
  if (w == 0) tdm_load_chunk(Wu_p, lds_offset(&sB[0][0]));
  for (int c = 0; c < 4; ++c) {
    if (w == 0) __builtin_amdgcn_s_wait_tensorcnt(0);
    __syncthreads();
    if (w == 0 && c + 1 < 4)
      tdm_load_chunk(Wu_p + (size_t)(c + 1) * 4096, lds_offset(&sB[(c + 1) & 1][0]));
    if (active) gemm_chunk(sB[c & 1], &sA[w][lo][0], c, hi, lane, acc);
  }
  if (active) {
#pragma unroll
    for (int nt = 0; nt < 8; ++nt) {
      const int col = nt * 16 + lo;
      const float bb = bu[col];
#pragma unroll
      for (int r = 0; r < 8; ++r)
        out[(size_t)(n0 + hi * 8 + r) * CIN + col] = acc[nt][r] + bb;
    }
  }
}

// =================== host side ===================
extern "C" void kernel_launch(void* const* d_in, const int* in_sizes, int n_in,
                              void* d_out, int out_size, void* d_ws, size_t ws_size,
                              hipStream_t stream) {
  const float* x   = (const float*)d_in[0];
  const int*   ei  = (const int*)d_in[1];
  const float* ea  = (const float*)d_in[2];
  const float* Wf  = (const float*)d_in[3];
  const float* bfv = (const float*)d_in[4];
  const float* Wq  = (const float*)d_in[5];
  const float* Wk  = (const float*)d_in[6];
  const float* Wa  = (const float*)d_in[7];
  const float* Wu  = (const float*)d_in[8];
  const float* bu  = (const float*)d_in[9];
  float* out = (float*)d_out;

  char* p = (char*)d_ws;
  auto take = [&](size_t bytes) {
    char* r = p;
    p += (bytes + 255) & ~(size_t)255;
    return r;
  };
  unsigned short* Wf_p = (unsigned short*)take((size_t)15 * 4096 * 2);
  unsigned short* Wq_p = (unsigned short*)take((size_t)4 * 4096 * 2);
  unsigned short* Wk_p = (unsigned short*)take((size_t)4 * 4096 * 2);
  unsigned short* Wu_p = (unsigned short*)take((size_t)4 * 4096 * 2);
  float* a_sum = (float*)take((size_t)NNODES * 4);
  float* a_ws  = (float*)take((size_t)NEDGES * 4);
  float* aggr  = (float*)take((size_t)NNODES * CIN * 4);
  unsigned int* f_ws = (unsigned int*)take((size_t)NEDGES * 64 * 4);

  zero_f32<<<dim3((NNODES + 255) / 256), dim3(256), 0, stream>>>(a_sum, NNODES);
  zero_f32<<<dim3((NNODES * CIN + 255) / 256), dim3(256), 0, stream>>>(aggr, NNODES * CIN);

  const int pack_elems = 15 * 4096 + 3 * 4 * 4096;  // 110592
  pack_weights_kernel<<<dim3((pack_elems + 255) / 256), dim3(256), 0, stream>>>(
      Wf, Wq, Wk, Wu, Wf_p, Wq_p, Wk_p, Wu_p);

  edge_pass_a<<<dim3(NEDGES / 64), dim3(128), 0, stream>>>(
      x, ei, ea, Wf_p, Wq_p, Wk_p, bfv, Wa, f_ws, a_ws, a_sum);

  edge_pass_b<<<dim3(NEDGES / 8), dim3(256), 0, stream>>>(ei, f_ws, a_ws, a_sum, aggr);

  node_pass<<<dim3((NNODES + 63) / 64), dim3(128), 0, stream>>>(x, aggr, Wu_p, bu, out);
}
